// NER_87531433492664
// MI455X (gfx1250) — matrix operand order repaired and checked by reading, not compiled, not run on hardware
//
#include <hip/hip_runtime.h>
#include <cstdint>
#include <cstddef>

typedef __attribute__((ext_vector_type(16))) __bf16 v16bf;
typedef __attribute__((ext_vector_type(8)))  float  v8f;
typedef __attribute__((ext_vector_type(4)))  unsigned int u32x4;

union ATile { v16bf v; u32x4 q[2]; };

// Padded sequence length: 2 zero rows + 256 + 2 zero rows per batch.
#define SPAD 260

__device__ __forceinline__ unsigned short f2bf(float x) {
  union { float f; unsigned int u; } c; c.f = x;
  unsigned int r = c.u + 0x7FFFu + ((c.u >> 16) & 1u);
  return (unsigned short)(r >> 16);
}

__device__ __forceinline__ float fast_sigmoid(float x) {
  // 1 / (1 + 2^(-x*log2(e)))  -> v_exp_f32 + v_rcp_f32
  return __builtin_amdgcn_rcpf(1.0f + __builtin_amdgcn_exp2f(-1.4426950408889634f * x));
}
__device__ __forceinline__ float fast_tanh(float x) {
  // 2*sigmoid(2x) - 1
  return 2.0f * __builtin_amdgcn_rcpf(1.0f + __builtin_amdgcn_exp2f(-2.8853900817779268f * x)) - 1.0f;
}

// ---------------------------------------------------------------------------
// 1) Embedding gather into zero-padded bf16 X: rows laid out b*260 + (s+2),
//    with rows s=-2,-1,256,257 zeroed so conv shifts never go out of bounds.
// ---------------------------------------------------------------------------
__global__ void k_embed(const int* __restrict__ tokens,
                        const float* __restrict__ char_emb,
                        const float* __restrict__ bigram_emb,
                        unsigned short* __restrict__ Xpad) {
  const int prow = blockIdx.x;          // 0..8*260-1
  const int t    = threadIdx.x;         // 0..255
  const int b    = prow / SPAD;
  const int s    = prow - b * SPAD - 2; // -2..257
  float v = 0.0f;
  if ((unsigned)s < 256u) {
    const int row = b * 256 + s;
    if (t < 128) {
      int tok = tokens[row * 3 + 0];
      v = char_emb[(size_t)tok * 128 + t];
    } else if (t < 192) {
      int tok = tokens[row * 3 + 1];
      v = bigram_emb[(size_t)tok * 64 + (t - 128)];
    } else {
      int tok = tokens[row * 3 + 2];
      v = bigram_emb[(size_t)tok * 64 + (t - 192)];
    }
  }
  Xpad[(size_t)prow * 256 + t] = f2bf(v);
}

// ---------------------------------------------------------------------------
// 2) Pack an [R x 256] f32 weight (R = KB*32 rows) into a WMMA bf16
//    B-operand layout, contiguous k as in the A loads: dst[((kb*16+nt)*32+lane)*8 + v] holds rows
//    (kb*32 + (lane&16?16:0) + 2v, +1) at column nt*16 + (lane&15).
// ---------------------------------------------------------------------------
__global__ void k_pack(const float* __restrict__ W, unsigned int* __restrict__ dst, int KB) {
  const int idx  = blockIdx.x * 256 + threadIdx.x;
  const int v    = idx & 7;
  const int lane = (idx >> 3) & 31;
  const int nt   = (idx >> 8) & 15;
  const int kb   = idx >> 12;
  if (kb >= KB) return;
  const int n  = (nt << 4) + (lane & 15);
  const int k0 = kb * 32 + ((lane & 16) ? 16 : 0) + 2 * v;
  unsigned short lo = f2bf(W[(size_t)k0 * 256 + n]);
  unsigned short hi = f2bf(W[(size_t)(k0 + 1) * 256 + n]);
  dst[idx] = (unsigned int)lo | ((unsigned int)hi << 16);
}

// ---------------------------------------------------------------------------
// 3) Fused conv1/conv3/conv5 (as shifted GEMMs) + bias + tanh + max.
//    One wave per 16x16 tile of C. 72 v_wmma_f32_16x16x32_bf16 per tile.
//    All A loads unconditional thanks to the padded X rows.
// ---------------------------------------------------------------------------
__global__ void __launch_bounds__(256) k_conv(
    const unsigned short* __restrict__ Xpad,
    const u32x4* __restrict__ w1pk,
    const u32x4* __restrict__ w3pk,
    const u32x4* __restrict__ w5pk,
    const float* __restrict__ cb1,
    const float* __restrict__ cb3,
    const float* __restrict__ cb5,
    float* __restrict__ C,
    unsigned short* __restrict__ Cbf) {
  const int lane = threadIdx.x & 31;
  const int wid  = threadIdx.x >> 5;
  const int tile = blockIdx.x * 8 + wid;   // 0..2047
  const int mt   = tile >> 4;              // 0..127 row-tile
  const int nt   = tile & 15;              // 0..15  col-tile
  const int row0 = mt << 4;                // global row base (b*256+s0)
  const int s0   = row0 & 255;
  const int b    = row0 >> 8;
  const int mrow = lane & 15;
  const int khalf = (lane & 16) ? 16 : 0;
  // padded row for this lane's M row at shift 0:
  const int prow0 = b * SPAD + 2 + s0 + mrow;

  v8f acc1 = {}; v8f acc3 = {}; v8f acc5 = {};

  for (int sh = -2; sh <= 2; ++sh) {
    const u32x4* arow =
        reinterpret_cast<const u32x4*>(Xpad + (size_t)(prow0 + sh) * 256);
    for (int kb = 0; kb < 8; ++kb) {
      const int qi = (kb * 32 + khalf) >> 3;
      ATile a;
      a.q[0] = arow[qi];
      a.q[1] = arow[qi + 1];
      {
        const int kbg = (sh + 2) * 8 + kb;
        const size_t bi = ((size_t)(kbg * 16 + nt) * 32 + lane) * 2;
        ATile bb; bb.q[0] = w5pk[bi]; bb.q[1] = w5pk[bi + 1];
        acc5 = __builtin_amdgcn_wmma_f32_16x16x32_bf16(false, a.v, false, bb.v,
                                                       (short)0, acc5, false, false);
      }
      if (sh >= -1 && sh <= 1) {
        const int kbg = (sh + 1) * 8 + kb;
        const size_t bi = ((size_t)(kbg * 16 + nt) * 32 + lane) * 2;
        ATile bb; bb.q[0] = w3pk[bi]; bb.q[1] = w3pk[bi + 1];
        acc3 = __builtin_amdgcn_wmma_f32_16x16x32_bf16(false, a.v, false, bb.v,
                                                       (short)0, acc3, false, false);
      }
      if (sh == 0) {
        const size_t bi = ((size_t)(kb * 16 + nt) * 32 + lane) * 2;
        ATile bb; bb.q[0] = w1pk[bi]; bb.q[1] = w1pk[bi + 1];
        acc1 = __builtin_amdgcn_wmma_f32_16x16x32_bf16(false, a.v, false, bb.v,
                                                       (short)0, acc1, false, false);
      }
    }
  }

  const int col = (nt << 4) + (lane & 15);
  const float bv1 = cb1[col], bv3 = cb3[col], bv5 = cb5[col];
  const int rbase = row0 + ((lane & 16) ? 8 : 0);
#pragma unroll
  for (int r = 0; r < 8; ++r) {
    float y = fmaxf(fast_tanh(acc1[r] + bv1),
                    fmaxf(fast_tanh(acc3[r] + bv3), fast_tanh(acc5[r] + bv5)));
    const size_t idx = (size_t)(rbase + r) * 256 + col;
    C[idx]   = y;
    Cbf[idx] = f2bf(y);
  }
}

// ---------------------------------------------------------------------------
// 4) g0 = C@W0 + b0, g1 = C@W1 + b1 in one pass (A fragment shared).
// ---------------------------------------------------------------------------
__global__ void __launch_bounds__(256) k_gemm2(
    const unsigned short* __restrict__ Cbf,
    const u32x4* __restrict__ W0pk, const u32x4* __restrict__ W1pk,
    const float* __restrict__ bias0, const float* __restrict__ bias1,
    float* __restrict__ g0, float* __restrict__ g1) {
  const int lane = threadIdx.x & 31;
  const int wid  = threadIdx.x >> 5;
  const int tile = blockIdx.x * 8 + wid;
  const int mt = tile >> 4;
  const int nt = tile & 15;
  const int row0 = mt << 4;
  const int mrow = lane & 15;
  const int khalf = (lane & 16) ? 16 : 0;

  v8f acc0 = {}; v8f accw1 = {};
  const u32x4* arow = reinterpret_cast<const u32x4*>(Cbf + (size_t)(row0 + mrow) * 256);
  for (int kb = 0; kb < 8; ++kb) {
    const int qi = (kb * 32 + khalf) >> 3;
    ATile a; a.q[0] = arow[qi]; a.q[1] = arow[qi + 1];
    const size_t bi = ((size_t)(kb * 16 + nt) * 32 + lane) * 2;
    ATile b0t; b0t.q[0] = W0pk[bi]; b0t.q[1] = W0pk[bi + 1];
    acc0 = __builtin_amdgcn_wmma_f32_16x16x32_bf16(false, a.v, false, b0t.v,
                                                   (short)0, acc0, false, false);
    ATile b1t; b1t.q[0] = W1pk[bi]; b1t.q[1] = W1pk[bi + 1];
    accw1 = __builtin_amdgcn_wmma_f32_16x16x32_bf16(false, a.v, false, b1t.v,
                                                    (short)0, accw1, false, false);
  }
  const int col = (nt << 4) + (lane & 15);
  const float bv0 = bias0[col], bv1 = bias1[col];
  const int rbase = row0 + ((lane & 16) ? 8 : 0);
#pragma unroll
  for (int r = 0; r < 8; ++r) {
    const size_t idx = (size_t)(rbase + r) * 256 + col;
    g0[idx] = acc0[r] + bv0;
    g1[idx] = accw1[r] + bv1;
  }
}

// ---------------------------------------------------------------------------
// 5) Pairwise sigmoid-gated masked mean + tanh.
//    Block = (b, 8 consecutive i); thread = h. Each g1/c load reused 8x.
// ---------------------------------------------------------------------------
__global__ void __launch_bounds__(256) k_pair(
    const float* __restrict__ g0, const float* __restrict__ g1,
    const float* __restrict__ C, const int* __restrict__ masks,
    float* __restrict__ rel) {
  const int b  = blockIdx.x >> 5;   // 0..7
  const int ig = blockIdx.x & 31;   // 0..31
  const int i0 = ig * 8;
  const int h  = threadIdx.x;       // 0..255
  const size_t base = (size_t)b * 256;

  float a0[8], mi[8], acc[8];
#pragma unroll
  for (int g = 0; g < 8; ++g) {
    a0[g]  = g0[(base + i0 + g) * 256 + h];
    mi[g]  = (float)masks[base + i0 + g];
    acc[g] = 0.0f;
  }
  float L = 0.0f;
  for (int j = 0; j < 256; ++j) {
    const float mj  = (float)masks[base + j];
    const float glv = g1[(base + j) * 256 + h];
    const float cj  = C[(base + j) * 256 + h];
    L += mj;
    const float w = mj * cj;
#pragma unroll
    for (int g = 0; g < 8; ++g)
      acc[g] += w * fast_sigmoid(a0[g] + glv);
  }
  const float invL = __builtin_amdgcn_rcpf(L);
#pragma unroll
  for (int g = 0; g < 8; ++g)
    rel[(base + i0 + g) * 256 + h] = fast_tanh(acc[g] * mi[g] * invL);
}

// ---------------------------------------------------------------------------
// 6) logits = rel @ Wo + bo ; softmax over 17 tags. One wave per row.
// ---------------------------------------------------------------------------
__global__ void k_out(const float* __restrict__ rel,
                      const float* __restrict__ Wo,
                      const float* __restrict__ bo,
                      float* __restrict__ out) {
  const int row  = blockIdx.x;    // 0..2047
  const int lane = threadIdx.x;   // 0..31
  __shared__ float r[256];
  __shared__ float lg[17];
  for (int i = lane; i < 256; i += 32) r[i] = rel[(size_t)row * 256 + i];
  __syncthreads();
  if (lane < 17) {
    float s = bo[lane];
    for (int k = 0; k < 256; ++k) s += r[k] * Wo[k * 17 + lane];
    lg[lane] = s;
  }
  __syncthreads();
  if (lane < 17) {
    float mx = lg[0];
    for (int k = 1; k < 17; ++k) mx = fmaxf(mx, lg[k]);
    float den = 0.0f;
    for (int k = 0; k < 17; ++k) den += __expf(lg[k] - mx);
    out[(size_t)row * 17 + lane] = __expf(lg[lane] - mx) / den;
  }
}

// ---------------------------------------------------------------------------
extern "C" void kernel_launch(void* const* d_in, const int* in_sizes, int n_in,
                              void* d_out, int out_size, void* d_ws, size_t ws_size,
                              hipStream_t stream) {
  (void)in_sizes; (void)n_in; (void)out_size; (void)ws_size;
  const int*   tokens     = (const int*)d_in[0];
  const int*   masks      = (const int*)d_in[1];
  const float* char_emb   = (const float*)d_in[2];
  const float* bigram_emb = (const float*)d_in[3];
  const float* conv1_w    = (const float*)d_in[4];
  const float* conv1_b    = (const float*)d_in[5];
  const float* conv3_w    = (const float*)d_in[6];
  const float* conv3_b    = (const float*)d_in[7];
  const float* conv5_w    = (const float*)d_in[8];
  const float* conv5_b    = (const float*)d_in[9];
  const float* W0         = (const float*)d_in[10];
  const float* b0         = (const float*)d_in[11];
  const float* W1         = (const float*)d_in[12];
  const float* b1         = (const float*)d_in[13];
  const float* Wo         = (const float*)d_in[14];
  const float* bo         = (const float*)d_in[15];

  char* p = (char*)d_ws;
  auto take = [&](size_t bytes) -> char* {
    char* r = p; p += (bytes + 255) & ~(size_t)255; return r;
  };
  unsigned short* Xpad = (unsigned short*)take((size_t)8 * SPAD * 256 * 2);
  unsigned int*   w1pk = (unsigned int*)take((size_t)8  * 4096 * 4);
  unsigned int*   w3pk = (unsigned int*)take((size_t)24 * 4096 * 4);
  unsigned int*   w5pk = (unsigned int*)take((size_t)40 * 4096 * 4);
  unsigned int*   W0pk = (unsigned int*)take((size_t)8  * 4096 * 4);
  unsigned int*   W1pk = (unsigned int*)take((size_t)8  * 4096 * 4);
  float*          Cmat = (float*)take((size_t)2048 * 256 * 4);
  unsigned short* Cbf  = (unsigned short*)take((size_t)2048 * 256 * 2);
  float*          g0   = (float*)take((size_t)2048 * 256 * 4);
  float*          g1   = (float*)take((size_t)2048 * 256 * 4);
  float*          rel  = (float*)take((size_t)2048 * 256 * 4);

  k_embed<<<dim3(8 * SPAD), dim3(256), 0, stream>>>(tokens, char_emb, bigram_emb, Xpad);

  k_pack<<<dim3(8  * 16), dim3(256), 0, stream>>>(conv1_w, w1pk, 8);
  k_pack<<<dim3(24 * 16), dim3(256), 0, stream>>>(conv3_w, w3pk, 24);
  k_pack<<<dim3(40 * 16), dim3(256), 0, stream>>>(conv5_w, w5pk, 40);
  k_pack<<<dim3(8  * 16), dim3(256), 0, stream>>>(W0, W0pk, 8);
  k_pack<<<dim3(8  * 16), dim3(256), 0, stream>>>(W1, W1pk, 8);

  k_conv<<<dim3(256), dim3(256), 0, stream>>>(
      Xpad, (const u32x4*)w1pk, (const u32x4*)w3pk, (const u32x4*)w5pk,
      conv1_b, conv3_b, conv5_b, Cmat, Cbf);

  k_gemm2<<<dim3(256), dim3(256), 0, stream>>>(
      Cbf, (const u32x4*)W0pk, (const u32x4*)W1pk, b0, b1, g0, g1);

  k_pair<<<dim3(256), dim3(256), 0, stream>>>(g0, g1, Cmat, masks, rel);

  k_out<<<dim3(2048), dim3(32), 0, stream>>>(rel, Wo, bo, (float*)d_out);
}
